// BaseLocalInference_82471962018650
// MI455X (gfx1250) — matrix operand order repaired and checked by reading, not compiled, not run on hardware
//
#include <hip/hip_runtime.h>
#include <hip/hip_bf16.h>
#include <math.h>

typedef _Float16 v8h  __attribute__((ext_vector_type(8)));
typedef _Float16 v16h __attribute__((ext_vector_type(16)));
typedef float    v8f  __attribute__((ext_vector_type(8)));

static constexpr int kLA = 4096;
static constexpr int kLB = 4096;
static constexpr int kD  = 1024;

// ---------------------------------------------------------------------------
// Fragment loaders using contiguous k order (K = kb + 16h + s) for both A and B
// A (16x32 f16): lanes 0-15 -> M=lane, K = kb+[0..15]
//                lanes 16-31 -> M=lane-16, K = kb+[16..31]
// -> two contiguous 16B chunks per lane.
__device__ __forceinline__ v16h load_a_frag(const _Float16* __restrict__ A,
                                            int row0, int kb, int ld, int lane) {
    int m = lane & 15, h = lane >> 4;
    const _Float16* p = A + (size_t)(row0 + m) * (size_t)ld + kb + h * 16;
    v8h lo = *(const v8h*)(p);
    v8h hi = *(const v8h*)(p + 8);
    return __builtin_shufflevector(lo, hi, 0,1,2,3,4,5,6,7,8,9,10,11,12,13,14,15);
}

// B (32x16 f16), operand provided as Bt = B^T (N x K row-major):
// lanes 0-15 -> N=lane, K = kb+[0..15]; lanes 16-31 -> N=lane-16, K = kb+[16..31]
// -> one contiguous 32B chunk per lane.
__device__ __forceinline__ v16h load_b_frag(const _Float16* __restrict__ Bt,
                                            int col0, int kb, int ld, int lane) {
    int n = lane & 15, h = lane >> 4;
    const _Float16* p = Bt + (size_t)(col0 + n) * (size_t)ld + kb + h * 16;
    return *(const v16h*)(p);
}

// ---------------------------------------------------------------------------
// Kernel 1: f32 -> f16 convert, producing row-major and transposed copies.
// Tile 32x32 through LDS so both writes are coalesced. grid.z selects a/b.
__global__ __launch_bounds__(256)
void convert_kernel(const float* __restrict__ a, const float* __restrict__ b,
                    _Float16* __restrict__ a16, _Float16* __restrict__ a16T,
                    _Float16* __restrict__ b16, _Float16* __restrict__ b16T) {
    const int R = kLA, C = kD;
    const float* src = (blockIdx.z == 0) ? a : b;
    _Float16* d0 = (blockIdx.z == 0) ? a16  : b16;
    _Float16* dT = (blockIdx.z == 0) ? a16T : b16T;

    __shared__ _Float16 s[32][33];
    int t  = threadIdx.x;
    int tx = t & 31, tg = t >> 5;
    int i0 = blockIdx.y * 32, j0 = blockIdx.x * 32;

#pragma unroll
    for (int q = 0; q < 4; ++q) {
        int r = tg + 8 * q;
        int i = i0 + r, j = j0 + tx;
        _Float16 h = (_Float16)src[(size_t)i * C + j];
        d0[(size_t)i * C + j] = h;
        s[tx][r] = h;                       // s[col_local][row_local]
    }
    __syncthreads();
#pragma unroll
    for (int q = 0; q < 4; ++q) {
        int x = tg + 8 * q;                 // local col -> row of transposed
        dT[(size_t)(j0 + x) * R + (i0 + tx)] = s[x][tx];
    }
}

// ---------------------------------------------------------------------------
// WMMA GEMM: C[M,N] = A[M,K] * Bt[N,K]^T, f16 in / f32 accumulate.
// 8 waves per WG arranged 4(M) x 2(N); wave tile 32x64 (2x4 WMMA 16x16x32);
// block tile 128(M) x 128(N). Register double-buffered K loop so the loads for
// step k+1 overlap the 8 WMMAs of step k; unconditional speculative prefetch.
// mode 0: store f32 C to Eout (ld = N).
// mode 1: fused m-vector epilogue: out4[{0,1,2,3}*M*N + i*N + j] = {x, w, x-w, x*w}
//         with x = base32[i*N + j], w = C[i][j].
__global__ __launch_bounds__(256)
void gemm_f16_kernel(const _Float16* __restrict__ A, const _Float16* __restrict__ Bt,
                     int K, int N, float* __restrict__ Eout, int mode,
                     const float* __restrict__ base32, float* __restrict__ out4) {
    int lane = threadIdx.x & 31;
    int w    = threadIdx.x >> 5;          // 0..7
    int wm   = w >> 1, wn = w & 1;        // 4 (M) x 2 (N) waves
    int row0 = blockIdx.y * 128 + wm * 32;
    int col0 = blockIdx.x * 128 + wn * 64;

    v8f acc[2][4];
#pragma unroll
    for (int mi = 0; mi < 2; ++mi)
#pragma unroll
        for (int ni = 0; ni < 4; ++ni) acc[mi][ni] = (v8f){};

    // per-lane base pointers for prefetch
    int am = lane & 15, ah = lane >> 4;
    const _Float16* pa0 = A + (size_t)(row0 + am)      * (size_t)K + ah * 8;
    const _Float16* pa1 = A + (size_t)(row0 + 16 + am) * (size_t)K + ah * 8;
    const _Float16* pb0 = Bt + (size_t)(col0 + am)      * (size_t)K + ah * 16;
    const _Float16* pb2 = Bt + (size_t)(col0 + 32 + am) * (size_t)K + ah * 16;

    // ---- preload K-step 0
    v16h a0 = load_a_frag(A, row0,      0, K, lane);
    v16h a1 = load_a_frag(A, row0 + 16, 0, K, lane);
    v16h b0 = load_b_frag(Bt, col0,      0, K, lane);
    v16h b1 = load_b_frag(Bt, col0 + 16, 0, K, lane);
    v16h b2 = load_b_frag(Bt, col0 + 32, 0, K, lane);
    v16h b3 = load_b_frag(Bt, col0 + 48, 0, K, lane);

    for (int kb = 32; kb < K; kb += 32) {
        // issue next step's loads first so they overlap this step's WMMAs
        v16h na0 = load_a_frag(A, row0,      kb, K, lane);
        v16h na1 = load_a_frag(A, row0 + 16, kb, K, lane);
        v16h nb0 = load_b_frag(Bt, col0,      kb, K, lane);
        v16h nb1 = load_b_frag(Bt, col0 + 16, kb, K, lane);
        v16h nb2 = load_b_frag(Bt, col0 + 32, kb, K, lane);
        v16h nb3 = load_b_frag(Bt, col0 + 48, kb, K, lane);
        // speculative prefetch one step further ahead (safe if OOB: dropped)
        __builtin_prefetch(pa0 + kb + 32, 0, 3);
        __builtin_prefetch(pa1 + kb + 32, 0, 3);
        __builtin_prefetch(pb0 + kb + 32, 0, 3);
        __builtin_prefetch(pb2 + kb + 32, 0, 3);

        acc[0][0] = __builtin_amdgcn_wmma_f32_16x16x32_f16(false, a0, false, b0, (short)0, acc[0][0], false, false);
        acc[0][1] = __builtin_amdgcn_wmma_f32_16x16x32_f16(false, a0, false, b1, (short)0, acc[0][1], false, false);
        acc[0][2] = __builtin_amdgcn_wmma_f32_16x16x32_f16(false, a0, false, b2, (short)0, acc[0][2], false, false);
        acc[0][3] = __builtin_amdgcn_wmma_f32_16x16x32_f16(false, a0, false, b3, (short)0, acc[0][3], false, false);
        acc[1][0] = __builtin_amdgcn_wmma_f32_16x16x32_f16(false, a1, false, b0, (short)0, acc[1][0], false, false);
        acc[1][1] = __builtin_amdgcn_wmma_f32_16x16x32_f16(false, a1, false, b1, (short)0, acc[1][1], false, false);
        acc[1][2] = __builtin_amdgcn_wmma_f32_16x16x32_f16(false, a1, false, b2, (short)0, acc[1][2], false, false);
        acc[1][3] = __builtin_amdgcn_wmma_f32_16x16x32_f16(false, a1, false, b3, (short)0, acc[1][3], false, false);

        a0 = na0; a1 = na1; b0 = nb0; b1 = nb1; b2 = nb2; b3 = nb3;
    }
    // final K-step
    acc[0][0] = __builtin_amdgcn_wmma_f32_16x16x32_f16(false, a0, false, b0, (short)0, acc[0][0], false, false);
    acc[0][1] = __builtin_amdgcn_wmma_f32_16x16x32_f16(false, a0, false, b1, (short)0, acc[0][1], false, false);
    acc[0][2] = __builtin_amdgcn_wmma_f32_16x16x32_f16(false, a0, false, b2, (short)0, acc[0][2], false, false);
    acc[0][3] = __builtin_amdgcn_wmma_f32_16x16x32_f16(false, a0, false, b3, (short)0, acc[0][3], false, false);
    acc[1][0] = __builtin_amdgcn_wmma_f32_16x16x32_f16(false, a1, false, b0, (short)0, acc[1][0], false, false);
    acc[1][1] = __builtin_amdgcn_wmma_f32_16x16x32_f16(false, a1, false, b1, (short)0, acc[1][1], false, false);
    acc[1][2] = __builtin_amdgcn_wmma_f32_16x16x32_f16(false, a1, false, b2, (short)0, acc[1][2], false, false);
    acc[1][3] = __builtin_amdgcn_wmma_f32_16x16x32_f16(false, a1, false, b3, (short)0, acc[1][3], false, false);

    // C/D layout: VGPR r, lanes 0-15 -> M=r, N=lane; lanes 16-31 -> M=8+r, N=lane-16
    int hh = lane >> 4, nn = lane & 15;
    size_t MN = (size_t)gridDim.y * 128 * (size_t)N;
#pragma unroll
    for (int mi = 0; mi < 2; ++mi)
#pragma unroll
        for (int ni = 0; ni < 4; ++ni) {
            v8f c = acc[mi][ni];
#pragma unroll
            for (int r = 0; r < 8; ++r) {
                int row = row0 + mi * 16 + hh * 8 + r;
                int col = col0 + ni * 16 + nn;
                size_t idx = (size_t)row * (size_t)N + col;
                if (mode == 0) {
                    Eout[idx] = c[r];
                } else {
                    float x = base32[idx];
                    float wv = c[r];
                    out4[idx]            = x;
                    out4[MN + idx]       = wv;
                    out4[2 * MN + idx]   = x - wv;
                    out4[3 * MN + idx]   = x * wv;
                }
            }
        }
}

// ---------------------------------------------------------------------------
// Kernel 3: per-row max and sum-exp of E (row length N=4096). One WG per row.
__global__ __launch_bounds__(256)
void rowstats_kernel(const float* __restrict__ E, float* __restrict__ mr,
                     float* __restrict__ sr) {
    const int N = kLB;
    int row = blockIdx.x, t = threadIdx.x;
    const float* p = E + (size_t)row * N;
    __shared__ float red[256];

    float m = -3.0e38f;
    for (int j = t; j < N; j += 256) m = fmaxf(m, p[j]);
    red[t] = m; __syncthreads();
    for (int s = 128; s > 0; s >>= 1) {
        if (t < s) red[t] = fmaxf(red[t], red[t + s]);
        __syncthreads();
    }
    float mrow = red[0]; __syncthreads();

    float s = 0.f;
    for (int j = t; j < N; j += 256) s += __expf(p[j] - mrow);
    red[t] = s; __syncthreads();
    for (int st = 128; st > 0; st >>= 1) {
        if (t < st) red[t] += red[t + st];
        __syncthreads();
    }
    if (t == 0) { mr[row] = mrow; sr[row] = red[0]; }
}

// Kernel 4: per-column max and sum-exp of E. 64 columns per WG, 4 row-stripes.
__global__ __launch_bounds__(256)
void colstats_kernel(const float* __restrict__ E, float* __restrict__ mc,
                     float* __restrict__ sc) {
    const int M = kLA, N = kLB;
    int tx = threadIdx.x & 63, ty = threadIdx.x >> 6;   // 64 cols x 4 stripes
    int j = blockIdx.x * 64 + tx;
    __shared__ float sm[4][64];
    __shared__ float cmax[64];

    float m = -3.0e38f;
    for (int i = ty; i < M; i += 4) m = fmaxf(m, E[(size_t)i * N + j]);
    sm[ty][tx] = m; __syncthreads();
    if (ty == 0) cmax[tx] = fmaxf(fmaxf(sm[0][tx], sm[1][tx]), fmaxf(sm[2][tx], sm[3][tx]));
    __syncthreads();
    float mm = cmax[tx];

    float s = 0.f;
    for (int i = ty; i < M; i += 4) s += __expf(E[(size_t)i * N + j] - mm);
    sm[ty][tx] = s; __syncthreads();
    if (ty == 0) {
        mc[j] = mm;
        sc[j] = sm[0][tx] + sm[1][tx] + sm[2][tx] + sm[3][tx];
    }
}

// Kernel 5: Pa[i][j] = exp(E-mr)/sr (f16), Pb[j][i] = exp(E-mc)/sc (f16),
// Pb written through an LDS 32x32 tile so both stores coalesce.
__global__ __launch_bounds__(256)
void softmax_kernel(const float* __restrict__ E,
                    const float* __restrict__ mr, const float* __restrict__ sr,
                    const float* __restrict__ mc, const float* __restrict__ sc,
                    _Float16* __restrict__ Pa, _Float16* __restrict__ Pb) {
    const int N = kLB;
    __shared__ _Float16 spb[32][33];
    int t  = threadIdx.x;
    int tx = t & 31, tg = t >> 5;
    int i0 = blockIdx.y * 32, j0 = blockIdx.x * 32;

#pragma unroll
    for (int q = 0; q < 4; ++q) {
        int i = i0 + tg + 8 * q;
        int j = j0 + tx;
        float e  = E[(size_t)i * N + j];
        float pa = __expf(e - mr[i]) / sr[i];
        float pb = __expf(e - mc[j]) / sc[j];
        Pa[(size_t)i * N + j] = (_Float16)pa;
        spb[tx][tg + 8 * q] = (_Float16)pb;   // spb[col_local][row_local]
    }
    __syncthreads();
#pragma unroll
    for (int q = 0; q < 4; ++q) {
        int x = tg + 8 * q;
        Pb[(size_t)(j0 + x) * kLA + (i0 + tx)] = spb[x][tx];
    }
}

// ---------------------------------------------------------------------------
extern "C" void kernel_launch(void* const* d_in, const int* in_sizes, int n_in,
                              void* d_out, int out_size, void* d_ws, size_t ws_size,
                              hipStream_t stream) {
    const float* a = (const float*)d_in[0];
    const float* b = (const float*)d_in[1];
    float* out = (float*)d_out;
    char* ws = (char*)d_ws;

    const size_t MB = 1024ull * 1024ull;
    _Float16* a16  = (_Float16*)(ws + 0);        //  8 MB: a as f16 [4096,1024]
    _Float16* b16  = (_Float16*)(ws + 8  * MB);  //  8 MB: b as f16 [4096,1024]
    _Float16* a16T = (_Float16*)(ws + 16 * MB);  //  8 MB: a^T f16 [1024,4096]
    _Float16* b16T = (_Float16*)(ws + 24 * MB);  //  8 MB: b^T f16 [1024,4096]
    float*    E    = (float*)   (ws + 32 * MB);  // 64 MB: energy f32 [4096,4096]
    _Float16* Pa   = (_Float16*)(ws + 96 * MB);  // 32 MB: row-softmax f16
    _Float16* Pb   = (_Float16*)(ws + 128 * MB); // 32 MB: col-softmax^T f16
    float* mr = (float*)(ws + 160 * MB);
    float* sr = mr + kLA;
    float* mc = sr + kLA;
    float* sc = mc + kLB;

    // 1. f16 conversion + transposed copies
    convert_kernel<<<dim3(kD / 32, kLA / 32, 2), 256, 0, stream>>>(
        a, b, a16, a16T, b16, b16T);

    // 2. E = a16 * b16^T  (M=4096, N=4096, K=1024), block tile 128x128
    gemm_f16_kernel<<<dim3(kLB / 128, kLA / 128), 256, 0, stream>>>(
        a16, b16, kD, kLB, E, 0, nullptr, nullptr);

    // 3/4. softmax statistics
    rowstats_kernel<<<kLA, 256, 0, stream>>>(E, mr, sr);
    colstats_kernel<<<kLB / 64, 256, 0, stream>>>(E, mc, sc);

    // 5. Pa / Pb (transposed) in f16
    softmax_kernel<<<dim3(kLB / 32, kLA / 32), 256, 0, stream>>>(
        E, mr, sr, mc, sc, Pa, Pb);

    // 6. wave_a = Pa * b16 fused with m_a epilogue (M=4096, N=1024, K=4096)
    gemm_f16_kernel<<<dim3(kD / 128, kLA / 128), 256, 0, stream>>>(
        Pa, b16T, kLB, kD, nullptr, 1, a, out);
    //    wave_b = Pb * a16 fused with m_b epilogue
    gemm_f16_kernel<<<dim3(kD / 128, kLB / 128), 256, 0, stream>>>(
        Pb, a16T, kLA, kD, nullptr, 1, b, out + 4ull * kLA * kD);
}
